// SequenceDecoder_11682311045453
// MI455X (gfx1250) — matrix operand order repaired and checked by reading, not compiled, not run on hardware
//
#include <hip/hip_runtime.h>

// ---------------------------------------------------------------------------
// GRU sequence decoder, fused persistent-tile kernel for gfx1250 (MI455X).
//   BATCH=32768, LATENT=HIDDEN=512, 3H=1536, SEQ=6, VOCAB=10
// Strategy: each workgroup owns a 32-row batch tile for the whole sequence.
//   gi (32x1536 bf16) and double-buffered h (2 x 32x512 bf16) live in LDS.
//   All GEMMs via v_wmma_f32_16x16x32_bf16; each wave computes the r/z/n gate
//   tiles of one 16x16 output tile together so the GRU pointwise math happens
//   on the accumulators in registers. Logits (VOCAB=10) done as VALU dots.
// ---------------------------------------------------------------------------

#define LATENT 512
#define HIDDEN 512
#define G3     1536
#define VOCAB  10
#define SEQ    6
#define BATCH  32768
#define MTILE  32
#define NWAVES 8

typedef __attribute__((ext_vector_type(16))) __bf16        v16bf;
typedef __attribute__((ext_vector_type(8)))  float         v8f;
typedef __attribute__((ext_vector_type(4)))  unsigned int  u32x4;

__device__ __forceinline__ unsigned short f2bf(float f) {
  unsigned u = __builtin_bit_cast(unsigned, f);
  u += 0x7FFFu + ((u >> 16) & 1u);              // round-to-nearest-even
  return (unsigned short)(u >> 16);
}
__device__ __forceinline__ float bf2f(unsigned short h) {
  unsigned u = ((unsigned)h) << 16;
  return __builtin_bit_cast(float, u);
}
__device__ __forceinline__ float sigm(float x) {
  return 1.0f / (1.0f + __expf(-x));
}

// -------- tiny pre-pass: convert w_ih / w_hh to bf16 into workspace --------
__global__ void cvt_weights_kernel(const float* __restrict__ wih,
                                   const float* __restrict__ whh,
                                   unsigned short* __restrict__ wih_bf,
                                   unsigned short* __restrict__ whh_bf, int n) {
  for (int i = blockIdx.x * blockDim.x + threadIdx.x; i < n;
       i += gridDim.x * blockDim.x) {
    wih_bf[i] = f2bf(wih[i]);
    whh_bf[i] = f2bf(whh[i]);
  }
}

// ------------------------------- fused GRU ---------------------------------
__global__ void __launch_bounds__(256)
gru_fused_kernel(const float* __restrict__ x,
                 const float* __restrict__ b_ih,
                 const float* __restrict__ b_hh,
                 const float* __restrict__ w_out,
                 const float* __restrict__ b_out,
                 const unsigned short* __restrict__ wih_bf,
                 const unsigned short* __restrict__ whh_bf,
                 float* __restrict__ out) {
  extern __shared__ char smem[];
  unsigned short* gi  = (unsigned short*)smem;          // 32*1536 bf16 = 96 KB
  unsigned short* hA  = gi + MTILE * G3;                // 32*512  bf16 = 32 KB
  unsigned short* hB  = hA + MTILE * HIDDEN;            // 32*512  bf16 = 32 KB
  float* wout_l       = (float*)(hB + MTILE * HIDDEN);  // 10*512 f32 = 20 KB
  float* bout_l       = wout_l + VOCAB * HIDDEN;        // 10 f32

  const int tid    = threadIdx.x;
  const int wave   = tid >> 5;
  const int lane   = tid & 31;
  const int col    = lane & 15;        // B-column / A-row / C-column in tile
  const int hi     = lane >> 4;        // lane-half selects contiguous K half
  const int batch0 = blockIdx.x * MTILE;

  // Stage small constants into LDS.
  for (int i = tid; i < VOCAB * HIDDEN; i += 256) wout_l[i] = w_out[i];
  if (tid < VOCAB) bout_l[tid] = b_out[tid];

  // Stage x tile -> hA (h0 = x), converting f32 -> bf16.
  for (int i = tid; i < MTILE * HIDDEN; i += 256) {
    hA[i] = f2bf(x[(size_t)(batch0 + (i >> 9)) * LATENT + (i & 511)]);
  }
  __syncthreads();

  // ---------------- gi = x @ w_ih^T + b_ih  (kept in LDS, bf16) ------------
  // 2 M-tiles x 96 N-tiles = 192 tiles, 24 per wave. K = 512 (16 wmma steps).
  for (int p = wave; p < 2 * 96; p += NWAVES) {
    const int m = p / 96, jt = p % 96;
    const int gbase = jt * 16;
    v8f acc = {};
    const unsigned short* wrow = wih_bf + (size_t)(gbase + col) * LATENT + hi * 16;
    const unsigned short* arow = hA + (m * 16 + col) * HIDDEN + hi * 16;
    for (int k = 0; k < LATENT / 32; ++k) {
      union { u32x4 q[2]; v16bf v; } au;                 // A: K{0-15}/{16-31}
      au.q[0] = *(const u32x4*)(arow + k * 32);
      au.q[1] = *(const u32x4*)(arow + k * 32 + 8);
      v16bf b = *(const v16bf*)(wrow + k * 32);          // B: 32 contiguous bytes
      acc = __builtin_amdgcn_wmma_f32_16x16x32_bf16(false, au.v, false, b,
                                                    (short)0, acc, false, false);
    }
    const float bias = b_ih[gbase + col];
    for (int i = 0; i < 8; ++i) {
      const int row = m * 16 + i + hi * 8;
      gi[row * G3 + gbase + col] = f2bf(acc[i] + bias);
    }
  }
  __syncthreads();

  unsigned short* hcur = hA;
  unsigned short* hnxt = hB;

  // ------------------------------ time loop --------------------------------
  for (int t = 0; t < SEQ; ++t) {
    // gh = h @ w_hh^T; each wave owns 8 (m,jt) tiles and computes all 3 gates.
    for (int p = wave; p < 2 * 32; p += NWAVES) {
      const int m = p >> 5, jt = p & 31;
      const int j16 = jt * 16;
      v8f ar = {}, az = {}, an = {};
      const unsigned short* arow = hcur + (m * 16 + col) * HIDDEN + hi * 16;
      const unsigned short* wr = whh_bf + (size_t)(j16 + col) * HIDDEN + hi * 16;
      const unsigned short* wz = wr + (size_t)HIDDEN * HIDDEN;   // +512 rows
      const unsigned short* wn = wz + (size_t)HIDDEN * HIDDEN;   // +1024 rows
      __builtin_prefetch(wr + HIDDEN, 0, 1);
      for (int k = 0; k < HIDDEN / 32; ++k) {
        union { u32x4 q[2]; v16bf v; } au;
        au.q[0] = *(const u32x4*)(arow + k * 32);
        au.q[1] = *(const u32x4*)(arow + k * 32 + 8);
        v16bf br = *(const v16bf*)(wr + k * 32);
        v16bf bz = *(const v16bf*)(wz + k * 32);
        v16bf bn = *(const v16bf*)(wn + k * 32);
        ar = __builtin_amdgcn_wmma_f32_16x16x32_bf16(false, au.v, false, br,
                                                     (short)0, ar, false, false);
        az = __builtin_amdgcn_wmma_f32_16x16x32_bf16(false, au.v, false, bz,
                                                     (short)0, az, false, false);
        an = __builtin_amdgcn_wmma_f32_16x16x32_bf16(false, au.v, false, bn,
                                                     (short)0, an, false, false);
      }
      // GRU pointwise on accumulators (C layout: lane half -> M = i or i+8).
      const int j = j16 + col;
      const float bhr = b_hh[j];
      const float bhz = b_hh[HIDDEN + j];
      const float bhn = b_hh[2 * HIDDEN + j];
      for (int i = 0; i < 8; ++i) {
        const int row = m * 16 + i + hi * 8;
        const float gir = bf2f(gi[row * G3 + j]);
        const float giz = bf2f(gi[row * G3 + HIDDEN + j]);
        const float gin = bf2f(gi[row * G3 + 2 * HIDDEN + j]);
        const float hold = bf2f(hcur[row * HIDDEN + j]);
        const float r = sigm(gir + ar[i] + bhr);
        const float z = sigm(giz + az[i] + bhz);
        const float n = tanhf(gin + r * (an[i] + bhn));
        const float hnew = (1.0f - z) * n + z * hold;
        hnxt[row * HIDDEN + j] = f2bf(hnew);
      }
    }
    __syncthreads();

    // logits[b, t, :] = h_new @ w_out^T + b_out   (VOCAB=10, pure VALU)
    for (int task = tid; task < MTILE * VOCAB; task += 256) {
      const int row = task / VOCAB, v = task % VOCAB;
      float sum = bout_l[v];
      const unsigned short* hr = hnxt + row * HIDDEN;
      const float* wv = wout_l + v * HIDDEN;
      for (int k = 0; k < HIDDEN; k += 2) {
        sum += bf2f(hr[k]) * wv[k] + bf2f(hr[k + 1]) * wv[k + 1];
      }
      out[(size_t)(batch0 + row) * (SEQ * VOCAB) + t * VOCAB + v] = sum;
    }

    unsigned short* tmp = hcur; hcur = hnxt; hnxt = tmp;
    __syncthreads();
  }
}

// ---------------------------------------------------------------------------
extern "C" void kernel_launch(void* const* d_in, const int* in_sizes, int n_in,
                              void* d_out, int out_size, void* d_ws, size_t ws_size,
                              hipStream_t stream) {
  (void)in_sizes; (void)n_in; (void)out_size; (void)ws_size;
  const float* x     = (const float*)d_in[0];
  const float* w_ih  = (const float*)d_in[1];
  const float* w_hh  = (const float*)d_in[2];
  const float* b_ih  = (const float*)d_in[3];
  const float* b_hh  = (const float*)d_in[4];
  const float* w_out = (const float*)d_in[5];
  const float* b_out = (const float*)d_in[6];
  float* out = (float*)d_out;

  // Workspace: bf16 copies of the two big weight matrices (~3.1 MB).
  unsigned short* wih_bf = (unsigned short*)d_ws;
  unsigned short* whh_bf = wih_bf + (size_t)G3 * LATENT;

  const int nw = G3 * LATENT;  // 786432 elements each
  cvt_weights_kernel<<<1024, 256, 0, stream>>>(w_ih, w_hh, wih_bf, whh_bf, nw);

  const size_t smem = (size_t)MTILE * G3 * 2          // gi bf16
                    + (size_t)2 * MTILE * HIDDEN * 2  // h double buffer bf16
                    + (size_t)VOCAB * HIDDEN * 4      // w_out f32
                    + 64;                             // b_out + pad
  gru_fused_kernel<<<BATCH / MTILE, 256, smem, stream>>>(
      x, b_ih, b_hh, w_out, b_out, wih_bf, whh_bf, out);
}
